// FaceRecogFC_74414603370896
// MI455X (gfx1250) — matrix operand order repaired and checked by reading, not compiled, not run on hardware
//
#include <hip/hip_runtime.h>
#include <hip/hip_bf16.h>

typedef __attribute__((ext_vector_type(16))) _Float16 v16h;
typedef __attribute__((ext_vector_type(8)))  _Float16 v8h;
typedef __attribute__((ext_vector_type(8)))  float    v8f;

#define BATCH 1024
#define EMB   512
#define NCLS  200000
#define NPAD  200064
#define NBLK  1563   // ceil(200000/128)

#if __has_builtin(__builtin_amdgcn_global_load_async_to_lds_b128)
#define HAS_ASYNC_LDS 1
typedef int v4i_vs __attribute__((vector_size(4 * sizeof(int))));
typedef __attribute__((address_space(1))) v4i_vs as1_v4i;
typedef __attribute__((address_space(3))) v4i_vs as3_v4i;
__device__ __forceinline__ void async_cp16(const _Float16* g, _Float16* l) {
  // GLOBAL_LOAD_ASYNC_TO_LDS_B128: per-lane 16B global -> LDS, ASYNCcnt tracked
  __builtin_amdgcn_global_load_async_to_lds_b128((as1_v4i*)g, (as3_v4i*)l, 0, 0);
}
__device__ __forceinline__ void wait_async0() {
#if __has_builtin(__builtin_amdgcn_s_wait_asynccnt)
  __builtin_amdgcn_s_wait_asynccnt(0);
#else
  asm volatile("s_wait_asynccnt 0" ::: "memory");
#endif
}
#endif

__device__ __forceinline__ float warp_sum32(float v) {
#pragma unroll
  for (int off = 16; off > 0; off >>= 1) v += __shfl_xor(v, off, 32);
  return v;
}

// ---------------------------------------------------------------- kernel A
// L2-normalize embedding rows, emit f16. One wave per row.
__global__ void k_norm_emb(const float* __restrict__ e, _Float16* __restrict__ neh) {
  const int wave = threadIdx.x >> 5, lane = threadIdx.x & 31;
  const int row = blockIdx.x * 8 + wave;
  float v[16];
  float ss = 0.f;
#pragma unroll
  for (int i = 0; i < 16; ++i) {
    v[i] = e[(size_t)row * EMB + i * 32 + lane];
    ss += v[i] * v[i];
  }
  ss = warp_sum32(ss);
  const float rn = 1.0f / fmaxf(sqrtf(ss), 1e-12f);
#pragma unroll
  for (int i = 0; i < 16; ++i)
    neh[(size_t)row * EMB + i * 32 + lane] = (_Float16)(v[i] * rn);
}

// ---------------------------------------------------------------- kernel B
// L2-normalize weight rows, emit f16; zero-fill pad rows. One wave per row.
__global__ void k_norm_w(const float* __restrict__ w, _Float16* __restrict__ nwh) {
  const int wave = threadIdx.x >> 5, lane = threadIdx.x & 31;
  const int row = blockIdx.x * 8 + wave;
  if (row >= NPAD) return;
  if (row >= NCLS) {
#pragma unroll
    for (int i = 0; i < 16; ++i)
      nwh[(size_t)row * EMB + i * 32 + lane] = (_Float16)0.0f;
    return;
  }
  float v[16];
  float ss = 0.f;
#pragma unroll
  for (int i = 0; i < 16; ++i) {
    v[i] = w[(size_t)row * EMB + i * 32 + lane];
    ss += v[i] * v[i];
  }
  ss = warp_sum32(ss);
  const float rn = 1.0f / fmaxf(sqrtf(ss), 1e-12f);
#pragma unroll
  for (int i = 0; i < 16; ++i)
    nwh[(size_t)row * EMB + i * 32 + lane] = (_Float16)(v[i] * rn);
}

// ---------------------------------------------------------------- kernel C
// Exact f32 target-class cosine per row. One wave per row.
__global__ void k_tdot(const float* __restrict__ e, const float* __restrict__ w,
                       const int* __restrict__ lab, float* __restrict__ td) {
  const int wave = threadIdx.x >> 5, lane = threadIdx.x & 31;
  const int row = blockIdx.x * 8 + wave;
  const int c = lab[row];
  float d = 0.f, se = 0.f, sw = 0.f;
#pragma unroll
  for (int i = 0; i < 16; ++i) {
    const int k = i * 32 + lane;
    const float a = e[(size_t)row * EMB + k];
    const float b = w[(size_t)c * EMB + k];
    d += a * b; se += a * a; sw += b * b;
  }
  d = warp_sum32(d); se = warp_sum32(se); sw = warp_sum32(sw);
  if (lane == 0)
    td[row] = d / (fmaxf(sqrtf(se), 1e-12f) * fmaxf(sqrtf(sw), 1e-12f));
}

// ---------------------------------------------------------------- kernel D
// Fused GEMM (WMMA f32_16x16x32_f16) + online softmax partials.
// Block tile 128x128, 256 threads = 8 waves arranged 2(M) x 4(N),
// each wave: 4x2 tiles of 16x16, K=512 in 16 steps, double-buffered LDS.
// Staging: GLOBAL_LOAD_ASYNC_TO_LDS_B128 when available (direct global->LDS,
// ASYNCcnt), else global->VGPR->LDS fallback.
__global__ __launch_bounds__(256) void k_gemm(const _Float16* __restrict__ A,
                                              const _Float16* __restrict__ B,
                                              float* __restrict__ pm,
                                              float* __restrict__ ps) {
  __shared__ __align__(64) _Float16 As[2][128][32];
  __shared__ __align__(64) _Float16 Bs[2][128][32];
  __shared__ float smM[4][128];
  __shared__ float smS[4][128];

  const int t = threadIdx.x, lane = t & 31, wave = t >> 5;
  const int wm = wave >> 2, wn = wave & 3;
  const int rowM0 = blockIdx.y * 128;
  const int colN0 = blockIdx.x * 128;

  v8f acc[4][2];
#pragma unroll
  for (int mt = 0; mt < 4; ++mt)
#pragma unroll
    for (int nt = 0; nt < 2; ++nt)
#pragma unroll
      for (int i = 0; i < 8; ++i) acc[mt][nt][i] = 0.f;

#if defined(HAS_ASYNC_LDS)
  // ---- async direct-to-LDS staging path ----
  {
#pragma unroll
    for (int j = 0; j < 2; ++j) {
      const int c = t + j * 256, r = c >> 2, o = (c & 3) * 8;
      async_cp16(&A[(size_t)(rowM0 + r) * EMB + o], &As[0][r][o]);
      async_cp16(&B[(size_t)(colN0 + r) * EMB + o], &Bs[0][r][o]);
    }
  }
  wait_async0();
  __syncthreads();
#else
  // ---- fallback: register staging ----
#pragma unroll
  for (int j = 0; j < 2; ++j) {
    const int c = t + j * 256, r = c >> 2, o = (c & 3) * 8;
    *(uint4*)&As[0][r][o] = *(const uint4*)&A[(size_t)(rowM0 + r) * EMB + o];
    *(uint4*)&Bs[0][r][o] = *(const uint4*)&B[(size_t)(colN0 + r) * EMB + o];
  }
  __syncthreads();
#endif

  for (int kt = 0; kt < 16; ++kt) {
    const int buf = kt & 1;

#if defined(HAS_ASYNC_LDS)
    if (kt < 15) {  // issue next stage straight into the other LDS buffer
      const int k0 = (kt + 1) * 32;
#pragma unroll
      for (int j = 0; j < 2; ++j) {
        const int c = t + j * 256, r = c >> 2, o = (c & 3) * 8;
        async_cp16(&A[(size_t)(rowM0 + r) * EMB + k0 + o], &As[buf ^ 1][r][o]);
        async_cp16(&B[(size_t)(colN0 + r) * EMB + k0 + o], &Bs[buf ^ 1][r][o]);
      }
    }
#else
    uint4 ra[2], rb[2];
    if (kt < 15) {
      const int k0 = (kt + 1) * 32;
#pragma unroll
      for (int j = 0; j < 2; ++j) {
        const int c = t + j * 256, r = c >> 2, o = (c & 3) * 8;
        ra[j] = *(const uint4*)&A[(size_t)(rowM0 + r) * EMB + k0 + o];
        rb[j] = *(const uint4*)&B[(size_t)(colN0 + r) * EMB + k0 + o];
      }
    }
    if (kt < 14) {  // emits global_prefetch_b8
      const int r = t >> 2, o = (t & 3) * 8, k2 = (kt + 2) * 32;
      __builtin_prefetch(&A[(size_t)(rowM0 + r) * EMB + k2 + o], 0, 0);
      __builtin_prefetch(&B[(size_t)(colN0 + r) * EMB + k2 + o], 0, 0);
    }
#endif

    // A-frag k order used here (contiguous, not the ISA table order): lanes 0-15 rows M=lane, K=0..15;
    // lanes 16-31 rows M=lane-16, K=16..31  -> one 32B LDS load/lane.
    // B-frag, same k order as A: lanes 0-15 col N=lane, K=0..15; lanes 16-31
    // col N=lane-16, K=16..31  -> one 32B LDS load/lane.
    const int ml = lane & 15, hg = lane >> 4;
    v16h af[4], bf[2];
#pragma unroll
    for (int mt = 0; mt < 4; ++mt) {
      const int m = wm * 64 + mt * 16 + ml;
      af[mt] = *(const v16h*)&As[buf][m][hg * 16];
    }
#pragma unroll
    for (int nt = 0; nt < 2; ++nt) {
      const int n = wn * 32 + nt * 16 + ml;
      bf[nt] = *(const v16h*)&Bs[buf][n][hg * 16];
    }
#pragma unroll
    for (int mt = 0; mt < 4; ++mt)
#pragma unroll
      for (int nt = 0; nt < 2; ++nt)
        acc[mt][nt] = __builtin_amdgcn_wmma_f32_16x16x32_f16(
            false, af[mt], false, bf[nt], (short)0, acc[mt][nt], false, false);

    if (kt < 15) {
#if defined(HAS_ASYNC_LDS)
      wait_async0();      // our async writes of stage kt+1 have landed in LDS
      __syncthreads();    // everyone's have
#else
#pragma unroll
      for (int j = 0; j < 2; ++j) {
        const int c = t + j * 256, r = c >> 2, o = (c & 3) * 8;
        *(uint4*)&As[buf ^ 1][r][o] = ra[j];
        *(uint4*)&Bs[buf ^ 1][r][o] = rb[j];
      }
      __syncthreads();
#endif
    }
  }

  // Epilogue: clamp*scale, per-row online (max, sumexp) over this block's cols.
  // C layout: VGPR v holds row v (lanes 0-15) / row v+8 (lanes 16-31), N=lane%16.
  const int ml = lane & 15, hg = lane >> 4;
  const int col0 = colN0 + wn * 32 + ml;
#pragma unroll
  for (int mt = 0; mt < 4; ++mt) {
    float m8[8], s8[8];
#pragma unroll
    for (int v = 0; v < 8; ++v) {
      float x0 = fminf(fmaxf(acc[mt][0][v], -1.f), 1.f) * 64.f;
      float x1 = fminf(fmaxf(acc[mt][1][v], -1.f), 1.f) * 64.f;
      if (col0 >= NCLS)      x0 = -1e30f;
      if (col0 + 16 >= NCLS) x1 = -1e30f;
      float m = fmaxf(x0, x1);
      float s = __expf(x0 - m) + __expf(x1 - m);
#pragma unroll
      for (int off = 1; off < 16; off <<= 1) {  // stays within 16-lane halves
        const float om = __shfl_xor(m, off, 32);
        const float os = __shfl_xor(s, off, 32);
        const float nm = fmaxf(m, om);
        s = s * __expf(m - nm) + os * __expf(om - nm);
        m = nm;
      }
      m8[v] = m; s8[v] = s;
    }
    if (ml == 0) {
      const int rbase = wm * 64 + mt * 16 + hg * 8;
#pragma unroll
      for (int v = 0; v < 8; ++v) { smM[wn][rbase + v] = m8[v]; smS[wn][rbase + v] = s8[v]; }
    }
  }
  __syncthreads();

  if (t < 128) {
    float M = smM[0][t], S = smS[0][t];
#pragma unroll
    for (int w2 = 1; w2 < 4; ++w2) {
      const float m2 = smM[w2][t], s2 = smS[w2][t];
      const float nm = fmaxf(M, m2);
      S = S * __expf(M - nm) + s2 * __expf(m2 - nm);
      M = nm;
    }
    const size_t idx = (size_t)blockIdx.x * BATCH + (rowM0 + t);
    pm[idx] = M; ps[idx] = S;
  }
}

// ---------------------------------------------------------------- kernel E
// Fold partials per row, apply exact arcface margin correction, mean-reduce.
__global__ void k_loss(const float* __restrict__ pm, const float* __restrict__ ps,
                       const float* __restrict__ td, float* __restrict__ out) {
  __shared__ float red[1024];
  const int t = threadIdx.x;
  float M = -3.0e38f, S = 0.f;
  for (int k = 0; k < NBLK; ++k) {
    const float m = pm[(size_t)k * BATCH + t];
    const float s = ps[(size_t)k * BATCH + t];
    const float nm = fmaxf(M, m);
    S = S * __expf(M - nm) + s * __expf(m - nm);
    M = nm;
  }
  const float tv    = td[t];
  const float lorig = 64.f * fminf(fmaxf(tv, -1.f), 1.f);
  const float tcc   = fminf(fmaxf(tv, -1.f + 1e-7f), 1.f - 1e-7f);
  const float ladj  = 64.f * cosf(acosf(tcc) + 0.5f);
  float S2 = S - __expf(lorig - M) + __expf(ladj - M);
  S2 = fmaxf(S2, 1e-30f);
  red[t] = (M + __logf(S2)) - ladj;
  __syncthreads();
  for (int s2 = 512; s2 > 0; s2 >>= 1) {
    if (t < s2) red[t] += red[t + s2];
    __syncthreads();
  }
  if (t == 0) out[0] = red[0] * (1.0f / BATCH);
}

// ---------------------------------------------------------------- launch
extern "C" void kernel_launch(void* const* d_in, const int* in_sizes, int n_in,
                              void* d_out, int out_size, void* d_ws, size_t ws_size,
                              hipStream_t stream) {
  const float* emb = (const float*)d_in[0];
  const float* wgt = (const float*)d_in[1];
  const int*   lab = (const int*)d_in[2];
  float* out = (float*)d_out;

  char* ws = (char*)d_ws;
  // ws layout (bytes), all 16B aligned:
  //   neh  : f16 [1024 x 512]            @ 0           (1,048,576)
  //   nwh  : f16 [200064 x 512]          @ 1,048,576   (204,865,536)
  //   pm   : f32 [1563 x 1024]           @ 205,914,112 (6,402,048)
  //   ps   : f32 [1563 x 1024]           @ 212,316,160 (6,402,048)
  //   td   : f32 [1024]                  @ 218,718,208 (4,096)
  _Float16* neh = (_Float16*)(ws + 0);
  _Float16* nwh = (_Float16*)(ws + 1048576);
  float*    pm  = (float*)(ws + 205914112);
  float*    ps  = (float*)(ws + 212316160);
  float*    td  = (float*)(ws + 218718208);

  k_norm_emb<<<128, 256, 0, stream>>>(emb, neh);
  k_norm_w<<<25008, 256, 0, stream>>>(wgt, nwh);
  k_tdot<<<128, 256, 0, stream>>>(emb, wgt, lab, td);
  k_gemm<<<dim3(NBLK, 8), 256, 0, stream>>>(neh, nwh, pm, ps);
  k_loss<<<1, 1024, 0, stream>>>(pm, ps, td, out);
}
